// YOLOv8PostProcessor_90263032693375
// MI455X (gfx1250) — matrix operand order repaired and checked by reading, not compiled, not run on hardware
//
#include <hip/hip_runtime.h>
#include <math.h>

#define NCLS     80
#define REGM     16
#define A_TOTAL  8400
#define NB       32
#define NCAND    1024
#define MAX_DET  300
#define IOU_THR  0.7f
#define CONF_THR 0.001f
#define MAX_WH   7680.0f

typedef __attribute__((ext_vector_type(16))) _Float16 v16h;
typedef __attribute__((ext_vector_type(8)))  float    v8f;

// Low 32 bits of a generic pointer into __shared__ memory == LDS byte offset
// (flat LDS aperture keeps the DS offset in addr[31:0], ISA §10.2).
__device__ __forceinline__ unsigned lds_addr32(const void* p) {
  return (unsigned)(unsigned long long)p;
}

// ---------------------------------------------------------------------------
// Kernel 1: fused decode. One block = 64 anchors of one image.
//   phase A: async-copy 64 dist channels x 64 anchors to LDS
//            (global_load_async_to_lds_b32, ASYNCcnt-fenced)
//   phase B: per wave, DFL via one v_wmma_f32_16x16x32_f16 (8 anchors/wave)
//   phase C: class max/argmax with coalesced global reads
// ---------------------------------------------------------------------------
__global__ __launch_bounds__(256)
void decode_kernel(const float* __restrict__ f0, const float* __restrict__ f1,
                   const float* __restrict__ f2,
                   float* __restrict__ boxes, float* __restrict__ confA,
                   float* __restrict__ clsA)
{
  __shared__ float sdist[64 * 64];     // [channel][anchor]
  __shared__ float sro[8 * 32 * 8];    // per-wave WMMA D spill
  __shared__ float pmax[64 * 4];
  __shared__ int   pidx[64 * 4];

  const int b    = blockIdx.y;
  const int base = blockIdx.x * 64;
  const int tid  = threadIdx.x;

  const float* feat; int HW, W, loff; float strideV;
  if (base < 6400)      { feat = f0; HW = 6400; W = 80; loff = 0;    strideV = 8.f;  }
  else if (base < 8000) { feat = f1; HW = 1600; W = 40; loff = 6400; strideV = 16.f; }
  else                  { feat = f2; HW = 400;  W = 20; loff = 8000; strideV = 32.f; }
  const float* fb = feat + (size_t)b * 144 * HW;

  // --- phase A: async stage distribution logits straight into LDS ---
  for (int i = tid; i < 64 * 64; i += 256) {
    int c = i >> 6, x = i & 63;
    int pos = base - loff + x;
    if (pos < HW) {
      unsigned ldst = lds_addr32(&sdist[c * 64 + x]);
      unsigned long long gsrc =
          (unsigned long long)(const void*)(fb + (size_t)c * HW + pos);
      asm volatile("global_load_async_to_lds_b32 %0, %1, off"
                   :: "v"(ldst), "v"(gsrc) : "memory");
    }
  }
  asm volatile("s_wait_asynccnt 0x0" ::: "memory");
  __syncthreads();

  // --- phase B: WMMA DFL ---
  const int wv   = tid >> 5;
  const int lane = tid & 31;
  {
    // A-matrix 16x32 f16 layout: lanes 0-15 -> row m, K{0..7}U{16..23};
    // lanes 16-31 -> row m, K{8..15}U{24..31}. Group0 = K0..15, Group1 = K16..31.
    int m  = lane & 15;            // row (anchor-side id & 15)
    int kb = (lane >> 4) * 8;      // bin base: 0 or 8
    int s  = m & 3;                // side (l,t,r,b)
    int a0 = m >> 2;               // group-0 anchor 0..3 (group-1 = a0+4)
    int x0 = wv * 8 + a0;
    int x1 = wv * 8 + 4 + a0;

    float l0[8], l1[8];
#pragma unroll
    for (int k = 0; k < 8; ++k) {
      int c = s * 16 + kb + k;     // dist channel = side*16 + bin
      l0[k] = sdist[c * 64 + x0];
      l1[k] = sdist[c * 64 + x1];
    }
    float m0 = l0[0], m1 = l1[0];
#pragma unroll
    for (int k = 1; k < 8; ++k) { m0 = fmaxf(m0, l0[k]); m1 = fmaxf(m1, l1[k]); }
    // full 16-bin row max: partner lane holds the other 8 bins
    m0 = fmaxf(m0, __shfl_xor(m0, 16));
    m1 = fmaxf(m1, __shfl_xor(m1, 16));

    v16h av;
#pragma unroll
    for (int k = 0; k < 8; ++k) {
      av[k]     = (_Float16)__expf(l0[k] - m0);   // group0 bins
      av[k + 8] = (_Float16)__expf(l1[k] - m1);   // group1 bins
    }

    // B 32x16 f16 (same K order as A): col N = lane & 15; elem k -> group k>>3, bin 8*(lane>>4)+(k&7).
    // col0 = proj (K<16), col1 = ones (K<16), col2 = proj (K>=16), col3 = ones.
    v16h bvv;
#pragma unroll
    for (int k = 0; k < 16; ++k) {
      int n = lane & 15, grp = k >> 3, bin = (lane >> 4) * 8 + (k & 7);
      float bvf = (n == 2 * grp) ? (float)bin
                : (n == 2 * grp + 1) ? 1.f : 0.f;
      bvv[k] = (_Float16)bvf;
    }

    v8f acc = {};
    acc = __builtin_amdgcn_wmma_f32_16x16x32_f16(false, av, false, bvv,
                                                 (short)0, acc, false, false);
    float* ro = &sro[wv * 256];
#pragma unroll
    for (int v = 0; v < 8; ++v) ro[lane * 8 + v] = acc[v];
  }
  __syncthreads();

  // readout: lane a (<8) owns anchor base + wv*8 + a
  if (lane < 8) {
    int a = lane;
    int g = a >> 2;
    const float* ro = &sro[wv * 256];
    float ltrb[4];
#pragma unroll
    for (int s = 0; s < 4; ++s) {
      int m  = (4 * a + s) & 15;
      int nn = 2 * g, nd = 2 * g + 1;
      // D(m,n): m<8 -> lane n, vgpr m ; m>=8 -> lane 16+n, vgpr m-8
      float num = (m < 8) ? ro[nn * 8 + m] : ro[(16 + nn) * 8 + (m - 8)];
      float den = (m < 8) ? ro[nd * 8 + m] : ro[(16 + nd) * 8 + (m - 8)];
      ltrb[s] = num * __builtin_amdgcn_rcpf(den);   // den >= 1 (softmax sum)
    }
    int anchor = base + wv * 8 + a;
    int pos = anchor - loff;
    if (pos < HW) {
      float ax = (float)(pos % W) + 0.5f;
      float ay = (float)(pos / W) + 0.5f;
      float cx = (ax + (ltrb[2] - ltrb[0]) * 0.5f) * strideV;
      float cy = (ay + (ltrb[3] - ltrb[1]) * 0.5f) * strideV;
      float w  = (ltrb[0] + ltrb[2]) * strideV;
      float h  = (ltrb[1] + ltrb[3]) * strideV;
      size_t o = ((size_t)b * A_TOTAL + anchor) * 4;
      boxes[o + 0] = cx; boxes[o + 1] = cy; boxes[o + 2] = w; boxes[o + 3] = h;
    }
  }

  // --- phase C: class max/argmax (sigmoid is monotone: compare logits) ---
  {
    int al = tid & 63;     // anchor within block (consecutive lanes -> coalesced)
    int p  = tid >> 6;     // class partition 0..3 (20 classes each)
    int anchor = base + al;
    int pos = anchor - loff;
    float mx = -1e30f; int mi = 0;
    if (pos < HW) {
      for (int j = 0; j < 20; ++j) {
        int c = p * 20 + j;
        float v = fb[(size_t)(64 + c) * HW + pos];
        if (v > mx) { mx = v; mi = c; }
      }
    }
    pmax[al * 4 + p] = mx;
    pidx[al * 4 + p] = mi;
  }
  __syncthreads();
  if (tid < 64) {
    int anchor = base + tid;
    int pos = anchor - loff;
    if (pos < HW) {
      float mx = pmax[tid * 4]; int mi = pidx[tid * 4];
#pragma unroll
      for (int p = 1; p < 4; ++p) {
        float v = pmax[tid * 4 + p];
        if (v > mx) { mx = v; mi = pidx[tid * 4 + p]; }
      }
      float conf = __builtin_amdgcn_rcpf(1.f + __expf(-mx));
      if (conf <= CONF_THR) conf = 0.f;
      size_t o = (size_t)b * A_TOTAL + anchor;
      confA[o] = conf;
      clsA[o]  = (float)mi;
    }
  }
}

// ---------------------------------------------------------------------------
// Kernel 2: per-image top-1024 via 2048-bin radix-histogram select.
// Candidate order is irrelevant: greedy NMS argmax imposes output order.
// ---------------------------------------------------------------------------
__global__ __launch_bounds__(256)
void topk_kernel(const float* __restrict__ boxes, const float* __restrict__ confA,
                 const float* __restrict__ clsA,
                 float* __restrict__ cbox, float* __restrict__ cconf,
                 float* __restrict__ ccls)
{
  __shared__ int hist[2048];
  __shared__ int s_cut, s_c1, s_n1, s_n2;
  const int b   = blockIdx.x;
  const int tid = threadIdx.x;
  const float* cf = confA + (size_t)b * A_TOTAL;

  for (int i = tid; i < 2048; i += 256) hist[i] = 0;
  if (tid == 0) { s_n1 = 0; s_n2 = 0; }
  for (int i = tid; i < NCAND; i += 256) {          // zero-pad candidates
    size_t so = (size_t)b * NCAND + i;
    cconf[so] = 0.f; ccls[so] = 0.f;
    cbox[so * 4 + 0] = 0.f; cbox[so * 4 + 1] = 0.f;
    cbox[so * 4 + 2] = 0.f; cbox[so * 4 + 3] = 0.f;
  }
  __syncthreads();

  for (int i = tid; i < A_TOTAL; i += 256) {
    unsigned key = __float_as_uint(cf[i]);          // conf >= 0: uint-ordered
    atomicAdd(&hist[key >> 21], 1);
  }
  __syncthreads();
  if (tid == 0) {
    int cum = 0, cut = 0, c1 = 0;
    for (int i = 2047; i >= 0; --i) {
      int h = hist[i];
      if (cum + h >= NCAND) { cut = i; c1 = cum; break; }
      cum += h;
    }
    s_cut = cut; s_c1 = c1;
  }
  __syncthreads();
  const int cut = s_cut, c1 = s_c1;
  for (int i = tid; i < A_TOTAL; i += 256) {
    float c = cf[i];
    int bin = (int)(__float_as_uint(c) >> 21);
    int slot = -1;
    if (bin > cut) slot = atomicAdd(&s_n1, 1);
    else if (bin == cut) {
      int r = atomicAdd(&s_n2, 1);
      if (r < NCAND - c1) slot = c1 + r;
    }
    if (slot >= 0) {
      size_t so = (size_t)b * NCAND + slot;
      size_t bo = ((size_t)b * A_TOTAL + i) * 4;
      cconf[so] = c;
      ccls[so]  = clsA[(size_t)b * A_TOTAL + i];
      cbox[so * 4 + 0] = boxes[bo + 0];
      cbox[so * 4 + 1] = boxes[bo + 1];
      cbox[so * 4 + 2] = boxes[bo + 2];
      cbox[so * 4 + 3] = boxes[bo + 3];
    }
  }
}

// ---------------------------------------------------------------------------
// Kernel 3: greedy NMS, one block per image. Scores + class-offset boxes live
// in registers (4 candidates/thread). 300 iters x {shfl argmax, suppress}.
// ---------------------------------------------------------------------------
__global__ __launch_bounds__(256)
void nms_kernel(const float* __restrict__ cbox, const float* __restrict__ cconf,
                const float* __restrict__ ccls, float* __restrict__ out)
{
  __shared__ float lbox[NCAND * 4];
  __shared__ float lcls[NCAND];
  __shared__ float wval[8];
  __shared__ int   wcid[8];
  __shared__ int   s_sel;
  __shared__ float selb[4];

  const int b    = blockIdx.x;
  const int tid  = threadIdx.x;
  const int lane = tid & 31;
  const int wv   = tid >> 5;

  for (int i = tid; i < NCAND; i += 256) {
    size_t so = (size_t)b * NCAND + i;
    lcls[i] = ccls[so];
    lbox[i * 4 + 0] = cbox[so * 4 + 0];
    lbox[i * 4 + 1] = cbox[so * 4 + 1];
    lbox[i * 4 + 2] = cbox[so * 4 + 2];
    lbox[i * 4 + 3] = cbox[so * 4 + 3];
  }
  __syncthreads();

  float sc[4], bx1[4], by1[4], bx2[4], by2[4];
#pragma unroll
  for (int i = 0; i < 4; ++i) {
    int cid = i * 256 + tid;
    sc[i] = cconf[(size_t)b * NCAND + cid];
    float cx = lbox[cid * 4 + 0], cy = lbox[cid * 4 + 1];
    float w  = lbox[cid * 4 + 2], h  = lbox[cid * 4 + 3];
    float off = lcls[cid] * MAX_WH;                 // per-class separation
    bx1[i] = cx - w * 0.5f + off;
    by1[i] = cy - h * 0.5f + off;
    bx2[i] = cx + w * 0.5f + off;
    by2[i] = cy + h * 0.5f + off;
  }

  for (int it = 0; it < MAX_DET; ++it) {
    // local argmax over 4 (ascending cid => strict > keeps lowest index)
    float v = sc[0]; int cid = tid;
#pragma unroll
    for (int i = 1; i < 4; ++i) {
      int c = i * 256 + tid;
      if (sc[i] > v) { v = sc[i]; cid = c; }
    }
    // wave32 reduction, lowest-index tie-break
#pragma unroll
    for (int off = 16; off > 0; off >>= 1) {
      float ov = __shfl_down(v, off);
      int   oc = __shfl_down(cid, off);
      if (ov > v || (ov == v && oc < cid)) { v = ov; cid = oc; }
    }
    if (lane == 0) { wval[wv] = v; wcid[wv] = cid; }
    __syncthreads();                                // sync A
    if (tid == 0) {
      float bv = wval[0]; int bc = wcid[0];
#pragma unroll
      for (int w = 1; w < 8; ++w)
        if (wval[w] > bv || (wval[w] == bv && wcid[w] < bc)) { bv = wval[w]; bc = wcid[w]; }
      s_sel = bc;
      float* orow = out + ((size_t)b * MAX_DET + it) * 6;
      if (bv > CONF_THR) {
        orow[0] = lbox[bc * 4 + 0]; orow[1] = lbox[bc * 4 + 1];
        orow[2] = lbox[bc * 4 + 2]; orow[3] = lbox[bc * 4 + 3];
        orow[4] = bv;               orow[5] = lcls[bc];
      } else {
        orow[0] = 0.f; orow[1] = 0.f; orow[2] = 0.f;
        orow[3] = 0.f; orow[4] = 0.f; orow[5] = 0.f;
      }
    }
    __syncthreads();                                // sync B
    const int sel = s_sel;
    if (tid == (sel & 255)) {
      int oi = sel >> 8;
      selb[0] = bx1[oi]; selb[1] = by1[oi]; selb[2] = bx2[oi]; selb[3] = by2[oi];
    }
    __syncthreads();                                // sync C
    const float sx1 = selb[0], sy1 = selb[1], sx2 = selb[2], sy2 = selb[3];
    const float sa = (sx2 - sx1) * (sy2 - sy1);
#pragma unroll
    for (int i = 0; i < 4; ++i) {
      float ix1 = fmaxf(sx1, bx1[i]);
      float iy1 = fmaxf(sy1, by1[i]);
      float ix2 = fminf(sx2, bx2[i]);
      float iy2 = fminf(sy2, by2[i]);
      float inter = fmaxf(ix2 - ix1, 0.f) * fmaxf(iy2 - iy1, 0.f);
      float aa = (bx2[i] - bx1[i]) * (by2[i] - by1[i]);
      float iou = inter / (sa + aa - inter + 1e-7f);
      if (iou > IOU_THR) sc[i] = 0.f;               // includes self (IoU==1)
    }
  }
}

// ---------------------------------------------------------------------------
extern "C" void kernel_launch(void* const* d_in, const int* in_sizes, int n_in,
                              void* d_out, int out_size, void* d_ws, size_t ws_size,
                              hipStream_t stream) {
  const float* f0 = (const float*)d_in[0];   // (32,144,80,80)
  const float* f1 = (const float*)d_in[1];   // (32,144,40,40)
  const float* f2 = (const float*)d_in[2];   // (32,144,20,20)
  float* out = (float*)d_out;                // (32,300,6)

  float* ws    = (float*)d_ws;               // ~7.3 MB of scratch
  float* boxes = ws;                                        // 32*8400*4
  float* confA = boxes + (size_t)NB * A_TOTAL * 4;          // 32*8400
  float* clsA  = confA + (size_t)NB * A_TOTAL;              // 32*8400
  float* cbox  = clsA  + (size_t)NB * A_TOTAL;              // 32*1024*4
  float* cconf = cbox  + (size_t)NB * NCAND * 4;            // 32*1024
  float* ccls  = cconf + (size_t)NB * NCAND;                // 32*1024

  dim3 gdec((A_TOTAL + 63) / 64, NB);
  decode_kernel<<<gdec, 256, 0, stream>>>(f0, f1, f2, boxes, confA, clsA);
  topk_kernel<<<NB, 256, 0, stream>>>(boxes, confA, clsA, cbox, cconf, ccls);
  nms_kernel<<<NB, 256, 0, stream>>>(cbox, cconf, ccls, out);
}
